// SimpleRNN_15951508537552
// MI455X (gfx1250) — matrix operand order repaired and checked by reading, not compiled, not run on hardware
//
#include <hip/hip_runtime.h>
#include <hip/hip_bf16.h>
#include <math.h>

// ---------------------------------------------------------------------------
// SimpleRNN for MI455X (gfx1250, wave32, WMMA):
//   xproj  : [32768 x 96] @ [96 x 1024]  bf16 WMMA, fp32 out
//   rnn    : 512 sequential steps of [64 x 1024] @ [1024 x 1024], persistent
//            grid, Whh column-slice resident in LDS, bf16 WMMA, tanh epilogue
//   logits : [32768 x 1024] @ [1024 x 96] bf16 WMMA, fp32 out
// ---------------------------------------------------------------------------

#define B_  64
#define T_  512
#define V_  96
#define H_  1024
#define M_  (B_ * T_)      // 32768 rows of x / hidden
#define RNN_WGS 32         // persistent workgroups for the recurrence

typedef __attribute__((ext_vector_type(16))) __bf16 v16bf;
typedef __attribute__((ext_vector_type(8)))  __bf16 v8bf;
typedef __attribute__((ext_vector_type(8)))  float  v8f;

// --- WMMA fragment loaders (ISA 7.12.2 layouts, wave32) --------------------

// A matrix 16x32 bf16 from row-major [rows x ld]:
// lane l<16 : row l,     K = {0..7, 16..23}
// lane l>=16: row l-16,  K = {8..15, 24..31}
static __device__ __forceinline__
v16bf frag_a_rowmajor(const __bf16* base, int row0, int k0, int ld, int lane) {
  const int l  = lane & 15;
  const int hi = (lane >> 4) & 1;
  const __bf16* p = base + (size_t)(row0 + l) * ld + k0 + hi * 8;
  v8bf lo  = *(const v8bf*)(p);
  v8bf hi8 = *(const v8bf*)(p + 16);
  return __builtin_shufflevector(lo, hi8, 0, 1, 2, 3, 4, 5, 6, 7,
                                 8, 9, 10, 11, 12, 13, 14, 15);
}

// B matrix 32x16 bf16 from K-major storage BT[n][k] (stride ld over k):
// lane l<16 : col l,     K = {0..7, 16..23}   lane l>=16: col l-16, K = {8..15, 24..31}
static __device__ __forceinline__
v16bf frag_b_kmajor(const __bf16* bt, int n0, int k0, int ld, int lane) {
  const int l  = lane & 15;
  const int hi = (lane >> 4) & 1;
  const __bf16* p = bt + (size_t)(n0 + l) * ld + k0 + hi * 8;
  v8bf lo  = *(const v8bf*)(p);
  v8bf hi8 = *(const v8bf*)(p + 16);
  return __builtin_shufflevector(lo, hi8, 0, 1, 2, 3, 4, 5, 6, 7,
                                 8, 9, 10, 11, 12, 13, 14, 15);
}

static __device__ __forceinline__ v8f wmma_bf16(v16bf a, v16bf b, v8f c) {
  return __builtin_amdgcn_wmma_f32_16x16x32_bf16(
      /*neg_a=*/false, a, /*neg_b=*/false, b,
      /*c_mod=*/(short)0, c, /*reuse_a=*/false, /*reuse_b=*/false);
}

// --- Kernel 0: pack/convert weights & activations, reset sync counter ------
__global__ void rnn_pack_kernel(const float* __restrict__ x,
                                const float* __restrict__ h0,
                                const float* __restrict__ Wxh,
                                const float* __restrict__ Whh,
                                const float* __restrict__ Wl,
                                __bf16* __restrict__ x_bf,
                                __bf16* __restrict__ hbuf0,
                                __bf16* __restrict__ WxhT,
                                __bf16* __restrict__ WhhT,
                                __bf16* __restrict__ Wl_bf,
                                unsigned* __restrict__ cnt) {
  const size_t tid    = (size_t)blockIdx.x * blockDim.x + threadIdx.x;
  const size_t stride = (size_t)gridDim.x * blockDim.x;
  if (tid == 0) *cnt = 0u;  // grid-sync counter; stream order makes this safe
  for (size_t i = tid; i < (size_t)M_ * V_; i += stride)
    x_bf[i] = (__bf16)x[i];
  for (size_t i = tid; i < (size_t)B_ * H_; i += stride)
    hbuf0[i] = (__bf16)h0[i];
  for (size_t i = tid; i < (size_t)H_ * V_; i += stride) {   // WxhT[h][v]
    const int h = (int)(i / V_), v = (int)(i % V_);
    WxhT[i] = (__bf16)Wxh[(size_t)v * H_ + h];
  }
  for (size_t i = tid; i < (size_t)H_ * H_; i += stride) {   // WhhT[n][k]
    const int n = (int)(i / H_), k = (int)(i % H_);
    WhhT[i] = (__bf16)Whh[(size_t)k * H_ + n];
  }
  for (size_t i = tid; i < (size_t)V_ * H_; i += stride)     // Wl is K-major
    Wl_bf[i] = (__bf16)Wl[i];
}

// --- Kernel 1: xproj = x @ Wxh + bh   (fp32 out, [M_ x H_]) ----------------
__global__ __launch_bounds__(128)
void rnn_xproj_kernel(const __bf16* __restrict__ x_bf,
                      const __bf16* __restrict__ WxhT,
                      const float* __restrict__ bh,
                      float* __restrict__ xproj) {
  const int lane = threadIdx.x & 31;
  const int wave = threadIdx.x >> 5;
  const int row0 = (blockIdx.x * 4 + wave) * 16;  // M tile
  const int n0   = blockIdx.y * 16;               // N tile
  v8f acc = {};
  #pragma unroll
  for (int kc = 0; kc < V_ / 32; ++kc) {          // K = 96 -> 3 WMMAs
    v16bf a = frag_a_rowmajor(x_bf, row0, kc * 32, V_, lane);
    v16bf b = frag_b_kmajor(WxhT, n0, kc * 32, V_, lane);
    acc = wmma_bf16(a, b, acc);
  }
  const int l  = lane & 15;
  const int hi = (lane >> 4) & 1;
  const int n  = n0 + l;
  const float bias = bh[n];
  #pragma unroll
  for (int r = 0; r < 8; ++r) {
    const int m = row0 + hi * 8 + r;
    xproj[(size_t)m * H_ + n] = acc[r] + bias;
  }
}

// --- Kernel 2: persistent recurrence h = tanh(xproj_t + h @ Whh) -----------
// 32 WGs x 8 waves; WG g owns H columns [g*32, g*32+32); that slice of WhhT
// lives in LDS for all 512 steps. Grid-wide release/acquire sync per step.
__global__ __launch_bounds__(256)
void rnn_recurrence_kernel(const __bf16* __restrict__ WhhT,
                           const float* __restrict__ xproj,
                           __bf16* __restrict__ hbufs,   // 2 x [B_ x H_]
                           __bf16* __restrict__ hidden,  // [M_ x H_] bf16
                           float* __restrict__ h_last,   // [B_ x H_] fp32
                           unsigned* __restrict__ cnt) {
  __shared__ __bf16 sB[32 * H_];  // 64 KB: 32 columns, K-major

  const int wg   = blockIdx.x;
  const int lane = threadIdx.x & 31;
  const int wave = threadIdx.x >> 5;

  // Stage Whh column slice (1024x32 bf16) into LDS once.
  {
    const v8bf* src = (const v8bf*)(WhhT + (size_t)wg * 32 * H_);
    v8bf* dst = (v8bf*)sB;
    for (int i = threadIdx.x; i < (32 * H_) / 8; i += blockDim.x) dst[i] = src[i];
  }
  __syncthreads();

  const int row0 = (wave & 3) * 16;              // M tile within B_=64
  const int nl0  = (wave >> 2) * 16;             // N tile within 32-col slice
  const int n0   = wg * 32 + nl0;                // global column base
  const int l    = lane & 15;
  const int hi   = (lane >> 4) & 1;
  const int n    = n0 + l;

  for (int t = 0; t < T_; ++t) {
    const __bf16* hprev = hbufs + (size_t)(t & 1) * (B_ * H_);
    __bf16* hnext       = hbufs + (size_t)((t + 1) & 1) * (B_ * H_);

    v8f acc = {};
    #pragma unroll 4
    for (int kc = 0; kc < H_ / 32; ++kc) {       // 32 chained bf16 WMMAs
      v16bf a = frag_a_rowmajor(hprev, row0, kc * 32, H_, lane);
      v16bf b = frag_b_kmajor(sB, nl0, kc * 32, H_, lane);  // ds_load_b128 x2
      acc = wmma_bf16(a, b, acc);
    }

    // Fused epilogue: + xproj, tanh, dual store (bf16 state + fp32 h_last).
    #pragma unroll
    for (int r = 0; r < 8; ++r) {
      const int m = row0 + hi * 8 + r;
      const size_t xoff = ((size_t)m * T_ + t) * H_ + n;
      const float v = tanhf(acc[r] + xproj[xoff]);
      hnext[(size_t)m * H_ + n] = (__bf16)v;
      hidden[xoff] = (__bf16)v;
      if (t == T_ - 1) {
        h_last[(size_t)m * H_ + n] = v;
      } else {
        __builtin_prefetch(&xproj[xoff + H_], 0, 1);  // global_prefetch_b8
      }
    }

    // Device-wide step barrier: release own writes, arrive, acquire-spin.
    __threadfence();
    __syncthreads();
    if (threadIdx.x == 0) {
      atomicAdd(cnt, 1u);
      const unsigned target = (unsigned)(t + 1) * RNN_WGS;
      while (__hip_atomic_load(cnt, __ATOMIC_ACQUIRE, __HIP_MEMORY_SCOPE_AGENT)
             < target) {
        __builtin_amdgcn_s_sleep(1);
      }
    }
    __syncthreads();
    __threadfence();  // acquire side for all threads before reading hprev
  }
}

// --- Kernel 3: logits = hidden @ Wl^T + bl  (fp32 out, [M_ x V_]) ----------
__global__ __launch_bounds__(128)
void rnn_logits_kernel(const __bf16* __restrict__ hidden,
                       const __bf16* __restrict__ Wl_bf,  // [V_ x H_] K-major
                       const float* __restrict__ bl,
                       float* __restrict__ out) {
  const int lane = threadIdx.x & 31;
  const int wave = threadIdx.x >> 5;
  const int row0 = (blockIdx.x * 4 + wave) * 16;  // M tile
  const int n0   = blockIdx.y * 16;               // N tile (V_=96 -> 6 tiles)
  v8f acc = {};
  #pragma unroll 4
  for (int kc = 0; kc < H_ / 32; ++kc) {
    v16bf a = frag_a_rowmajor(hidden, row0, kc * 32, H_, lane);
    v16bf b = frag_b_kmajor(Wl_bf, n0, kc * 32, H_, lane);
    acc = wmma_bf16(a, b, acc);
  }
  const int l  = lane & 15;
  const int hi = (lane >> 4) & 1;
  const int n  = n0 + l;
  const float bias = bl[n];
  #pragma unroll
  for (int r = 0; r < 8; ++r) {
    const int m = row0 + hi * 8 + r;
    out[(size_t)m * V_ + n] = acc[r] + bias;
  }
}

// ---------------------------------------------------------------------------
extern "C" void kernel_launch(void* const* d_in, const int* in_sizes, int n_in,
                              void* d_out, int out_size, void* d_ws, size_t ws_size,
                              hipStream_t stream) {
  const float* x   = (const float*)d_in[0];   // [B,T,V]
  const float* h0  = (const float*)d_in[1];   // [B,H]
  const float* Wxh = (const float*)d_in[2];   // [V,H]
  const float* Whh = (const float*)d_in[3];   // [H,H]
  const float* bh  = (const float*)d_in[4];   // [H]
  const float* Wl  = (const float*)d_in[5];   // [V,H]
  const float* bl  = (const float*)d_in[6];   // [V]

  float* out_logits = (float*)d_out;                       // [M_, V_]
  float* out_hlast  = (float*)d_out + (size_t)M_ * V_;     // [B_, H_]

  // Workspace carving (256B-aligned regions).
  char* ws = (char*)d_ws;
  size_t off = 0;
  auto take = [&](size_t bytes) -> void* {
    void* p = ws + off;
    off += (bytes + 255) & ~(size_t)255;
    return p;
  };
  float*   xproj  = (float*)take((size_t)M_ * H_ * sizeof(float));
  __bf16*  hidden = (__bf16*)take((size_t)M_ * H_ * sizeof(__bf16));
  __bf16*  hbufs  = (__bf16*)take((size_t)2 * B_ * H_ * sizeof(__bf16));
  __bf16*  WhhT   = (__bf16*)take((size_t)H_ * H_ * sizeof(__bf16));
  __bf16*  WxhT   = (__bf16*)take((size_t)H_ * V_ * sizeof(__bf16));
  __bf16*  Wl_bf  = (__bf16*)take((size_t)V_ * H_ * sizeof(__bf16));
  __bf16*  x_bf   = (__bf16*)take((size_t)M_ * V_ * sizeof(__bf16));
  unsigned* cnt   = (unsigned*)take(256);

  // 0) pack/convert (also zeroes the grid-sync counter, stream-ordered)
  rnn_pack_kernel<<<2048, 256, 0, stream>>>(x, h0, Wxh, Whh, Wl,
                                            x_bf, hbufs, WxhT, WhhT, Wl_bf, cnt);

  // 1) xproj: M tiles 2048 (4 waves/block), N tiles 64
  rnn_xproj_kernel<<<dim3(M_ / 16 / 4, H_ / 16), 128, 0, stream>>>(
      x_bf, WxhT, bh, xproj);

  // 2) persistent recurrence: 32 co-resident WGs, one 32-col Whh slice each
  rnn_recurrence_kernel<<<RNN_WGS, 256, 0, stream>>>(
      WhhT, xproj, hbufs, hidden, out_hlast, cnt);

  // 3) logits: M tiles 2048 (4 waves/block), N tiles 6
  rnn_logits_kernel<<<dim3(M_ / 16 / 4, V_ / 16), 128, 0, stream>>>(
      hidden, Wl_bf, bl, out_logits);
}
